// ViTEncoder_66082366816750
// MI455X (gfx1250) — matrix operand order repaired and checked by reading, not compiled, not run on hardware
//
#include <hip/hip_runtime.h>
#include <hip/hip_bf16.h>
#include <math.h>

typedef __bf16 bf16;
typedef __attribute__((ext_vector_type(16))) bf16  v16bf;
typedef __attribute__((ext_vector_type(8)))  bf16  v8bf;
typedef __attribute__((ext_vector_type(8)))  float v8f;
typedef int v4i_vs __attribute__((vector_size(16)));   // matches builtin's V4i

constexpr int BATCH = 4096;
constexpr int SEQ   = 14;
constexpr int DIM   = 768;
constexpr int NH    = 12;
constexpr int DH    = 64;
constexpr int DFF   = 3072;
constexpr int NT    = BATCH * SEQ;   // 57344 tokens

#if defined(__has_builtin)
#if __has_builtin(__builtin_amdgcn_global_load_async_to_lds_b128)
#define ASYNC_LDS 1
#endif
#endif

// 16-byte global -> LDS copy; async (ASYNCcnt) when available.
__device__ __forceinline__ void ld_b128(const bf16* g, bf16* l) {
#ifdef ASYNC_LDS
  __builtin_amdgcn_global_load_async_to_lds_b128(
      (__attribute__((address_space(1))) v4i_vs*)g,
      (__attribute__((address_space(3))) v4i_vs*)l, 0, 0);
#else
  *(v8bf*)l = *(const v8bf*)g;
#endif
}

template <int N>
__device__ __forceinline__ void wait_async() {
#ifdef ASYNC_LDS
#if __has_builtin(__builtin_amdgcn_s_wait_asynccnt)
  __builtin_amdgcn_s_wait_asynccnt(N);
#else
  asm volatile("s_wait_asynccnt %0" :: "n"(N) : "memory");
#endif
#endif
}

__device__ __forceinline__ v16bf mk16(v8bf lo, v8bf hi) {
  v16bf r;
#pragma unroll
  for (int t = 0; t < 8; ++t) { r[t] = lo[t]; r[t + 8] = hi[t]; }
  return r;
}

// ---------------------------------------------------------------- utilities
__global__ __launch_bounds__(256)
void cvt_f32_to_bf16(const float* __restrict__ in, bf16* __restrict__ out, size_t n) {
  size_t i = (size_t)blockIdx.x * 256 + threadIdx.x;
  if (i < n) out[i] = (bf16)in[i];
}

// in[K][N] f32 -> out[N][K] bf16 (weights pre-transposed for WMMA B operand)
__global__ __launch_bounds__(256)
void cvt_transpose_bf16(const float* __restrict__ in, bf16* __restrict__ out,
                        int K, int N) {
  int k = blockIdx.x * 16 + (threadIdx.x & 15);
  int n = blockIdx.y * 16 + (threadIdx.x >> 4);
  out[(size_t)n * K + k] = (bf16)in[(size_t)k * N + n];
}

// per-head transpose: in[NH][DH][DH] -> out[NH][DH][DH] with [n][k] layout
__global__ __launch_bounds__(256)
void cvt_transpose_head(const float* __restrict__ in, bf16* __restrict__ out) {
  int hh = blockIdx.x;
  for (int idx = threadIdx.x; idx < DH * DH; idx += 256) {
    int k = idx >> 6, n = idx & 63;
    out[hh * DH * DH + n * DH + k] = (bf16)in[hh * DH * DH + k * DH + n];
  }
}

__global__ __launch_bounds__(256)
void pe_init(float* __restrict__ pe) {
  int idx = blockIdx.x * 256 + threadIdx.x;
  if (idx >= SEQ * (DIM / 2)) return;
  int s = idx / (DIM / 2), i = idx % (DIM / 2);
  float term = expf(-(float)(2 * i) * (logf(10000.0f) / (float)DIM));
  float a = (float)s * term;
  pe[s * DIM + 2 * i]     = sinf(a);
  pe[s * DIM + 2 * i + 1] = cosf(a);
}

// ------------------------------------------------------------- layernorm
__global__ __launch_bounds__(256)
void ln_bf16(const float* __restrict__ x, const float* __restrict__ g,
             const float* __restrict__ be, bf16* __restrict__ out) {
  __shared__ float red[256];
  size_t row = blockIdx.x;
  const float* xr = x + row * DIM;
  float lx[3];
  float s = 0.f;
#pragma unroll
  for (int i = 0; i < 3; ++i) { lx[i] = xr[threadIdx.x + i * 256]; s += lx[i]; }
  red[threadIdx.x] = s; __syncthreads();
  for (int st = 128; st > 0; st >>= 1) {
    if (threadIdx.x < st) red[threadIdx.x] += red[threadIdx.x + st];
    __syncthreads();
  }
  float mu = red[0] * (1.0f / DIM);
  __syncthreads();
  float v = 0.f;
#pragma unroll
  for (int i = 0; i < 3; ++i) { float d = lx[i] - mu; v += d * d; }
  red[threadIdx.x] = v; __syncthreads();
  for (int st = 128; st > 0; st >>= 1) {
    if (threadIdx.x < st) red[threadIdx.x] += red[threadIdx.x + st];
    __syncthreads();
  }
  float rs = rsqrtf(red[0] * (1.0f / DIM) + 1e-5f);
#pragma unroll
  for (int i = 0; i < 3; ++i) {
    int c = threadIdx.x + i * 256;
    out[row * DIM + c] = (bf16)((lx[i] - mu) * rs * g[c] + be[c]);
  }
}

// ------------------------------------------------- generic WMMA bf16 GEMM
// A: [M][K] bf16, Bt: [N][K] bf16 (pre-transposed weights)
// EPI 0: out_f32 = 2*acc + 2*bias + pe[(m%S)][n]      (patch embed + pos)
// EPI 1: out_f32 = acc + bias + res                    (residual add)
// EPI 2: out_bf16 = relu(acc + bias)                   (FFN hidden)
template <int EPI>
__global__ __launch_bounds__(256)
void gemm_bf16_wmma(const bf16* __restrict__ A, const bf16* __restrict__ Bt,
                    int N, int K,
                    const float* __restrict__ bias,
                    const float* __restrict__ res,
                    const float* __restrict__ pe,
                    float* __restrict__ outF, bf16* __restrict__ outH) {
  constexpr int BM = 128, BN = 128, BK = 32;
  constexpr int LDT = 40;   // padded bf16 stride (80 B rows, 16B-aligned chunks)
  __shared__ __align__(16) bf16 sA[2][BM * LDT];
  __shared__ __align__(16) bf16 sB[2][BN * LDT];   // [n][k]
  const int tid  = threadIdx.x;
  const int wave = tid >> 5, lane = tid & 31;
  const int l16 = lane & 15, lh = lane >> 4;
  const int m0 = blockIdx.x * BM, n0 = blockIdx.y * BN;
  const int wm = (wave & 3) * 32, wn = (wave >> 2) * 64;
  v8f acc[2][4] = {};
  const int arow = tid >> 1, akp = (tid & 1) * 16;   // 128 rows x two 16B chunks

  auto issue_tile = [&](int kb, int buf) {
    const bf16* asrc = A  + (size_t)(m0 + arow) * K + kb + akp;
    const bf16* bsrc = Bt + (size_t)(n0 + arow) * K + kb + akp;
    ld_b128(asrc,     &sA[buf][arow * LDT + akp]);
    ld_b128(asrc + 8, &sA[buf][arow * LDT + akp + 8]);
    ld_b128(bsrc,     &sB[buf][arow * LDT + akp]);
    ld_b128(bsrc + 8, &sB[buf][arow * LDT + akp + 8]);
  };

  const int nk = K / BK;
  issue_tile(0, 0);
  for (int t = 0; t < nk; ++t) {
    if (t + 1 < nk) { issue_tile((t + 1) * BK, (t + 1) & 1); wait_async<4>(); }
    else            { wait_async<0>(); }
    __syncthreads();
    const bf16* cA = sA[t & 1];
    const bf16* cB = sB[t & 1];
    v16bf af[2], bfr[4];
#pragma unroll
    for (int i = 0; i < 2; ++i) {   // A frag: row=l16, K chunks {lh*8, lh*8+16}
      int row = wm + i * 16 + l16;
      af[i] = mk16(*(const v8bf*)&cA[row * LDT + lh * 8],
                   *(const v8bf*)&cA[row * LDT + lh * 8 + 16]);
    }
#pragma unroll
    for (int j = 0; j < 4; ++j) {   // B frag: col=l16, K chunks {lh*8, lh*8+16}
      int col = wn + j * 16 + l16;
      bfr[j] = mk16(*(const v8bf*)&cB[col * LDT + lh * 8],
                    *(const v8bf*)&cB[col * LDT + lh * 8 + 16]);
    }
#pragma unroll
    for (int i = 0; i < 2; ++i)
#pragma unroll
      for (int j = 0; j < 4; ++j)
        acc[i][j] = __builtin_amdgcn_wmma_f32_16x16x32_bf16(
            false, af[i], false, bfr[j], (short)0, acc[i][j], false, false);
    __syncthreads();
  }
  // epilogue: D layout — lane<16: M=r, lane>=16: M=8+r; N = l16
#pragma unroll
  for (int i = 0; i < 2; ++i)
#pragma unroll
    for (int j = 0; j < 4; ++j)
#pragma unroll
      for (int r = 0; r < 8; ++r) {
        int m = m0 + wm + i * 16 + r + lh * 8;
        int n = n0 + wn + j * 16 + l16;
        size_t o = (size_t)m * N + n;
        float v = acc[i][j][r];
        if (EPI == 0) {
          outF[o] = 2.0f * v + 2.0f * bias[n] + pe[(m % SEQ) * N + n];
        } else if (EPI == 1) {
          outF[o] = v + bias[n] + res[o];
        } else {
          float t2 = v + bias[n];
          outH[o] = (bf16)(t2 > 0.0f ? t2 : 0.0f);
        }
      }
}

// --------------------------------------------- fused per-(b,h) attention
// weights arrive pre-transposed per head: wT[h][n][k]
__global__ __launch_bounds__(32)
void attn_fused(const bf16* __restrict__ h1,
                const bf16* __restrict__ wqT, const bf16* __restrict__ wkT,
                const bf16* __restrict__ wvT,
                const float* __restrict__ bq, const float* __restrict__ bk,
                const float* __restrict__ bv,
                bf16* __restrict__ attn) {
  constexpr int LDW = 72, LDP = 40, LDV = 40;
  __shared__ __align__(16) bf16 sX[16 * LDW];
  __shared__ __align__(16) bf16 sW[3][DH * LDW];   // [n][k]
  __shared__ __align__(16) bf16 sQ[16 * LDW];      // [s][k]
  __shared__ __align__(16) bf16 sK[16 * LDW];      // [t][k]
  __shared__ __align__(16) bf16 sVT[DH * LDV];     // [e][t], t padded to 32
  __shared__ __align__(16) bf16 sP[16 * LDP];      // [s][t], t padded to 32
  __shared__ float sS[16 * 16];

  const int b = blockIdx.x, hh = blockIdx.y;
  const int lane = threadIdx.x;
  const int l16 = lane & 15, lh = lane >> 4;

  // async-stage X rows (s < SEQ): 14 rows x four 16B chunks
  for (int idx = lane; idx < SEQ * 4; idx += 32) {
    int s = idx >> 2, q = (idx & 3) * 16;
    ld_b128(h1 + ((size_t)(b * SEQ + s)) * DIM + hh * DH + q, &sX[s * LDW + q]);
  }
  // async-stage the three transposed weight matrices: 64 rows x 4 chunks each
  const bf16* wsrc[3] = { wqT + hh * DH * DH, wkT + hh * DH * DH, wvT + hh * DH * DH };
  for (int w = 0; w < 3; ++w)
    for (int idx = lane; idx < DH * 4; idx += 32) {
      int n = idx >> 2, q = (idx & 3) * 16;
      ld_b128(wsrc[w] + n * DH + q, &sW[w][n * LDW + q]);
    }
  // zero padding: X rows 14..15, V^T, P
  for (int idx = lane; idx < 2 * DH; idx += 32) {
    int s = 14 + (idx >> 6), k = idx & 63;
    sX[s * LDW + k] = (bf16)0.0f;
  }
  for (int idx = lane; idx < DH * LDV; idx += 32) sVT[idx] = (bf16)0.0f;
  for (int idx = lane; idx < 16 * LDP; idx += 32) sP[idx] = (bf16)0.0f;
  wait_async<0>();
  __syncthreads();

  // Q/K/V = Xs @ W + b   (M=16, N=64, K=64)
  const float* bias[3] = { bq + hh * DH, bk + hh * DH, bv + hh * DH };
  for (int w = 0; w < 3; ++w) {
    v8f acc[4] = {};
#pragma unroll
    for (int ks = 0; ks < 2; ++ks) {
      int kb = ks * 32;
      v16bf af = mk16(*(const v8bf*)&sX[l16 * LDW + kb + lh * 8],
                      *(const v8bf*)&sX[l16 * LDW + kb + lh * 8 + 16]);
#pragma unroll
      for (int j = 0; j < 4; ++j) {
        int col = j * 16 + l16;
        v16bf bfv = mk16(*(const v8bf*)&sW[w][col * LDW + kb + lh * 8],
                         *(const v8bf*)&sW[w][col * LDW + kb + lh * 8 + 16]);
        acc[j] = __builtin_amdgcn_wmma_f32_16x16x32_bf16(
            false, af, false, bfv, (short)0, acc[j], false, false);
      }
    }
#pragma unroll
    for (int j = 0; j < 4; ++j)
#pragma unroll
      for (int r = 0; r < 8; ++r) {
        int m = r + lh * 8;
        int n = j * 16 + l16;
        float v = acc[j][r] + bias[w][n];
        if (w == 0)      sQ[m * LDW + n]  = (bf16)v;
        else if (w == 1) sK[m * LDW + n]  = (bf16)v;
        else             sVT[n * LDV + m] = (bf16)v;   // store V transposed
      }
    __syncthreads();
  }

  // scores = Q @ K^T * (1/8)
  v8f sc = {};
#pragma unroll
  for (int ks = 0; ks < 2; ++ks) {
    int kb = ks * 32;
    v16bf af = mk16(*(const v8bf*)&sQ[l16 * LDW + kb + lh * 8],
                    *(const v8bf*)&sQ[l16 * LDW + kb + lh * 8 + 16]);
    v16bf bfv = mk16(*(const v8bf*)&sK[l16 * LDW + kb + lh * 8],
                     *(const v8bf*)&sK[l16 * LDW + kb + lh * 8 + 16]);
    sc = __builtin_amdgcn_wmma_f32_16x16x32_bf16(
        false, af, false, bfv, (short)0, sc, false, false);
  }
#pragma unroll
  for (int r = 0; r < 8; ++r) sS[(r + lh * 8) * 16 + l16] = sc[r] * 0.125f;
  __syncthreads();

  // softmax over valid SEQ columns; padded rows/cols stay zero in sP
  if (lane < 16) {
    int row = lane;
    if (row < SEQ) {
      float mx = -1e30f;
      for (int t = 0; t < SEQ; ++t) mx = fmaxf(mx, sS[row * 16 + t]);
      float e[SEQ];
      float sum = 0.f;
      for (int t = 0; t < SEQ; ++t) { e[t] = expf(sS[row * 16 + t] - mx); sum += e[t]; }
      float inv = 1.0f / sum;
      for (int t = 0; t < SEQ; ++t) sP[row * LDP + t] = (bf16)(e[t] * inv);
    }
  }
  __syncthreads();

  // out = P @ V   (M=16, N=64, K=32 with zero padding)
  v16bf ap = mk16(*(const v8bf*)&sP[l16 * LDP + lh * 8],
                  *(const v8bf*)&sP[l16 * LDP + lh * 8 + 16]);
#pragma unroll
  for (int j = 0; j < 4; ++j) {
    int col = j * 16 + l16;
    v16bf bv16 = mk16(*(const v8bf*)&sVT[col * LDV + lh * 8],
                      *(const v8bf*)&sVT[col * LDV + lh * 8 + 16]);
    v8f ao = __builtin_amdgcn_wmma_f32_16x16x32_bf16(
        false, ap, false, bv16, (short)0, v8f{}, false, false);
#pragma unroll
    for (int r = 0; r < 8; ++r) {
      int m = r + lh * 8;
      if (m < SEQ)
        attn[((size_t)(b * SEQ + m)) * DIM + hh * DH + col] = (bf16)ao[r];
    }
  }
}

// ----------------------------------------------- mean-pool + classifier
__global__ __launch_bounds__(256)
void head_cls(const float* __restrict__ enc, const float* __restrict__ Wc,
              const float* __restrict__ bc, float* __restrict__ out) {
  __shared__ float r0[256], r1[256];
  int b = blockIdx.x;
  float a0 = 0.f, a1 = 0.f;
  for (int d = threadIdx.x; d < DIM; d += 256) {
    float m = 0.f;
    for (int s = 0; s < SEQ; ++s) m += enc[((size_t)(b * SEQ + s)) * DIM + d];
    m *= (1.0f / SEQ);
    a0 += m * Wc[d * 2 + 0];
    a1 += m * Wc[d * 2 + 1];
  }
  r0[threadIdx.x] = a0; r1[threadIdx.x] = a1;
  __syncthreads();
  for (int st = 128; st > 0; st >>= 1) {
    if (threadIdx.x < st) {
      r0[threadIdx.x] += r0[threadIdx.x + st];
      r1[threadIdx.x] += r1[threadIdx.x + st];
    }
    __syncthreads();
  }
  if (threadIdx.x == 0) {
    out[b * 2 + 0] = r0[0] + bc[0];
    out[b * 2 + 1] = r1[0] + bc[1];
  }
}

// ------------------------------------------------------------------ launch
extern "C" void kernel_launch(void* const* d_in, const int* in_sizes, int n_in,
                              void* d_out, int out_size, void* d_ws, size_t ws_size,
                              hipStream_t stream) {
  (void)in_sizes; (void)n_in; (void)out_size; (void)ws_size;
  const float* x   = (const float*)d_in[0];
  const float* Wpe = (const float*)d_in[1];
  const float* bpe = (const float*)d_in[2];
  const float* Wq  = (const float*)d_in[3];
  const float* bq  = (const float*)d_in[4];
  const float* Wk  = (const float*)d_in[5];
  const float* bk  = (const float*)d_in[6];
  const float* Wv  = (const float*)d_in[7];
  const float* bv  = (const float*)d_in[8];
  const float* Wo  = (const float*)d_in[9];
  const float* bo  = (const float*)d_in[10];
  const float* g1  = (const float*)d_in[11];
  const float* be1 = (const float*)d_in[12];
  const float* g2  = (const float*)d_in[13];
  const float* be2 = (const float*)d_in[14];
  const float* W1  = (const float*)d_in[15];
  const float* b1  = (const float*)d_in[16];
  const float* W2  = (const float*)d_in[17];
  const float* b2  = (const float*)d_in[18];
  const float* Wc  = (const float*)d_in[19];
  const float* bc  = (const float*)d_in[20];
  float* out = (float*)d_out;

  char* base = (char*)d_ws;
  size_t off = 0;
  auto alloc = [&](size_t bytes) -> void* {
    void* p = base + off;
    off += (bytes + 255) & ~(size_t)255;
    return p;
  };
  float* peb = (float*)alloc((size_t)SEQ * DIM * 4);
  bf16*  xb  = (bf16*) alloc((size_t)NT * DIM * 2);   // x bf16, reused for attn out
  bf16*  h12 = (bf16*) alloc((size_t)NT * DIM * 2);   // LN1 out, then LN2 out
  float* xwp = (float*)alloc((size_t)NT * DIM * 4);   // x_with_pos, reused for enc
  float* x2  = (float*)alloc((size_t)NT * DIM * 4);
  bf16*  ff1 = (bf16*) alloc((size_t)NT * DFF * 2);
  bf16* wpeT = (bf16*) alloc((size_t)DIM * DIM * 2);
  bf16* wqT  = (bf16*) alloc((size_t)NH * DH * DH * 2);
  bf16* wkT  = (bf16*) alloc((size_t)NH * DH * DH * 2);
  bf16* wvT  = (bf16*) alloc((size_t)NH * DH * DH * 2);
  bf16* woT  = (bf16*) alloc((size_t)DIM * DIM * 2);
  bf16* w1T  = (bf16*) alloc((size_t)DIM * DFF * 2);  // [DFF][DIM]
  bf16* w2T  = (bf16*) alloc((size_t)DFF * DIM * 2);  // [DIM][DFF]

  pe_init<<<(SEQ * (DIM / 2) + 255) / 256, 256, 0, stream>>>(peb);
  cvt_f32_to_bf16<<<(unsigned)(((size_t)NT * DIM + 255) / 256), 256, 0, stream>>>(
      x, xb, (size_t)NT * DIM);
  cvt_transpose_bf16<<<dim3(DIM / 16, DIM / 16), 256, 0, stream>>>(Wpe, wpeT, DIM, DIM);
  cvt_transpose_bf16<<<dim3(DIM / 16, DIM / 16), 256, 0, stream>>>(Wo,  woT,  DIM, DIM);
  cvt_transpose_bf16<<<dim3(DIM / 16, DFF / 16), 256, 0, stream>>>(W1,  w1T,  DIM, DFF);
  cvt_transpose_bf16<<<dim3(DFF / 16, DIM / 16), 256, 0, stream>>>(W2,  w2T,  DFF, DIM);
  cvt_transpose_head<<<NH, 256, 0, stream>>>(Wq, wqT);
  cvt_transpose_head<<<NH, 256, 0, stream>>>(Wk, wkT);
  cvt_transpose_head<<<NH, 256, 0, stream>>>(Wv, wvT);

  // x_with_pos = 2*(x@Wpe) + 2*b_pe + pe
  gemm_bf16_wmma<0><<<dim3(NT / 128, DIM / 128), 256, 0, stream>>>(
      xb, wpeT, DIM, DIM, bpe, nullptr, peb, xwp, nullptr);
  ln_bf16<<<NT, 256, 0, stream>>>(xwp, g1, be1, h12);
  attn_fused<<<dim3(BATCH, NH), 32, 0, stream>>>(h12, wqT, wkT, wvT, bq, bk, bv, xb);
  // x2 = x_with_pos + attn@Wo + bo
  gemm_bf16_wmma<1><<<dim3(NT / 128, DIM / 128), 256, 0, stream>>>(
      xb, woT, DIM, DIM, bo, xwp, nullptr, x2, nullptr);
  ln_bf16<<<NT, 256, 0, stream>>>(x2, g2, be2, h12);
  // ff1 = relu(h2@W1 + b1)
  gemm_bf16_wmma<2><<<dim3(NT / 128, DFF / 128), 256, 0, stream>>>(
      h12, w1T, DFF, DIM, b1, nullptr, nullptr, nullptr, ff1);
  // enc = x2 + ff1@W2 + b2
  gemm_bf16_wmma<1><<<dim3(NT / 128, DIM / 128), 256, 0, stream>>>(
      ff1, w2T, DIM, DFF, b2, x2, nullptr, xwp, nullptr);
  head_cls<<<BATCH, 256, 0, stream>>>(xwp, Wc, bc, out);
}
